// DeepConvNet_47957604827234
// MI455X (gfx1250) — matrix operand order repaired and checked by reading, not compiled, not run on hardware
//
#include <hip/hip_runtime.h>

// ---------------------------------------------------------------------------
// DeepConvNet forward for MI455X (gfx1250).
// bf16 WMMA implicit-GEMM convs with fused bias+ReLU+2x2 maxpool, activations
// kept as bf16 NHWC so halo tiles stream into LDS via async-to-LDS copies.
// ---------------------------------------------------------------------------

typedef __attribute__((ext_vector_type(16))) __bf16          v16bf;
typedef __attribute__((ext_vector_type(8)))  float           v8f;
typedef __attribute__((ext_vector_type(8)))  unsigned short  v8us;

union Frag16 {
  v16bf v;
  v8us  h[2];
  unsigned short u[16];
};

__device__ __forceinline__ unsigned short f2bf(float f) {
  unsigned int u = __float_as_uint(f);
  u += 0x7FFFu + ((u >> 16) & 1u);   // round-to-nearest-even
  return (unsigned short)(u >> 16);
}
__device__ __forceinline__ float bf2f(unsigned short s) {
  return __uint_as_float(((unsigned int)s) << 16);
}

// Pack weights (F,CIN,3,3) fp32 -> bf16 [F][Kpad], K ordered (ky*3+kx)*CIN + c,
// zero-padded to Kpad (multiple of 32) so the WMMA K-loop has no tail.
__global__ void prep_weights_kernel(const float* __restrict__ w,
                                    unsigned short* __restrict__ wp,
                                    int F, int CIN, int Kpad) {
  int idx = blockIdx.x * blockDim.x + threadIdx.x;
  int total = F * Kpad;
  if (idx >= total) return;
  int f = idx / Kpad;
  int k = idx - f * Kpad;
  int K = CIN * 9;
  float v = 0.0f;
  if (k < K) {
    int pos = k / CIN;          // ky*3 + kx
    int c   = k - pos * CIN;
    int ky  = pos / 3;
    int kx  = pos - ky * 3;
    v = w[((f * CIN + c) * 3 + ky) * 3 + kx];
  }
  wp[idx] = f2bf(v);
}

// ---------------------------------------------------------------------------
// Conv3x3 (pad 1) + bias + ReLU + 2x2 maxpool, fused.
// One wave: two 16(filter) x 16(x) WMMA tiles for rows y0,y0+1 sharing the A
// fragment, then in-register 2x2 pooling; writes pooled bf16 NHWC.
// LDS halo: [4 rows][18 cols][CIN] bf16. BF16IN selects input layout:
//   false -> fp32 NCHW (network input), true -> bf16 NHWC (pooled activations).
// grid = (HW/16, HW/2, 64 * F/(16*WAVES)), block = 32*WAVES.
// ---------------------------------------------------------------------------
template <int CIN, int F, int HW, int WAVES, bool BF16IN>
__global__ __launch_bounds__(32 * WAVES)
void conv_pool_wmma(const void* __restrict__ Xv,
                    const unsigned short* __restrict__ Wp,
                    const float* __restrict__ bias,
                    unsigned short* __restrict__ P) {
  constexpr int K     = CIN * 9;
  constexpr int Kpad  = (K + 31) & ~31;
  constexpr int NFB   = F / (16 * WAVES);
  constexpr int STAGE = 4 * 18 * CIN;
  constexpr int HO    = HW / 2;

  __shared__ unsigned short smem[STAGE];   // [ky 0..3][xx 0..17][c], bf16

  const int lane = threadIdx.x & 31;
  const int wave = threadIdx.x >> 5;
  const int m    = lane & 15;   // A row (filter) / B col (x offset)
  const int half = lane >> 4;

  const int xt = blockIdx.x * 16;
  const int yo = blockIdx.y;          // pooled output row
  const int y0 = yo * 2;              // first conv row
  const int n  = blockIdx.z / NFB;
  const int f0 = (blockIdx.z % NFB) * (16 * WAVES) + wave * 16;

  // ---- Stage halo tile: rows y0-1 .. y0+2, cols xt-1 .. xt+16, all channels.
  if constexpr (BF16IN) {
    // bf16 NHWC input: each (ky,xx) pixel is CIN contiguous bf16 -> 16B runs.
    const unsigned short* Xn = (const unsigned short*)Xv +
                               (size_t)n * HW * HW * CIN;
    const unsigned long long sbase = (unsigned long long)(const void*)Xn;
    constexpr int NV = 4 * 18 * (CIN / 8);
    for (int i = threadIdx.x; i < NV; i += 32 * WAVES) {
      int cv = i % (CIN / 8);
      int t  = i / (CIN / 8);
      int xx = t % 18;
      int ky = t / 18;
      int gy = y0 + ky - 1;
      int gx = xt + xx - 1;
      unsigned dstE = (unsigned)((ky * 18 + xx) * CIN + cv * 8);
      if (gy >= 0 && gy < HW && gx >= 0 && gx < HW) {
        unsigned soff   = (unsigned)((((unsigned)(gy * HW + gx)) * CIN + cv * 8) * 2);
        unsigned ldsoff = (unsigned)(size_t)(const void*)(smem + dstE);
        // CDNA5 async copy: LDS[vdst] = MEM[saddr + vaddr], tracked by ASYNCcnt.
        asm volatile("global_load_async_to_lds_b128 %0, %1, %2"
                     :: "v"(ldsoff), "v"(soff), "s"(sbase)
                     : "memory");
      } else {
        v8us z = {0, 0, 0, 0, 0, 0, 0, 0};
        *(v8us*)(smem + dstE) = z;
      }
    }
    asm volatile("s_wait_asynccnt 0x0" ::: "memory");
  } else {
    // fp32 NCHW network input (CIN=3): scalar convert path.
    const float* Xn = (const float*)Xv;
    for (int idx = threadIdx.x; idx < STAGE; idx += 32 * WAVES) {
      int c  = idx % CIN;
      int t  = idx / CIN;
      int xx = t % 18;
      int ky = t / 18;
      int gy = y0 + ky - 1;
      int gx = xt + xx - 1;
      float v = 0.0f;
      if (gy >= 0 && gy < HW && gx >= 0 && gx < HW)
        v = Xn[(((size_t)n * CIN + c) * HW + gy) * HW + gx];
      smem[idx] = f2bf(v);
    }
  }
  __syncthreads();

  const unsigned short* wrow = Wp + (size_t)(f0 + m) * Kpad;

  v8f accA = {0.f, 0.f, 0.f, 0.f, 0.f, 0.f, 0.f, 0.f};
  v8f accB = {0.f, 0.f, 0.f, 0.f, 0.f, 0.f, 0.f, 0.f};

  for (int kb = 0; kb < Kpad; kb += 32) {
    Frag16 a, b0, b1;
    // A (16x32 bf16): lane K = kb + 16*half + e -> one contiguous
    // 16-element run of the packed weight row.
    a.h[0] = *(const v8us*)(wrow + kb + 16 * half);
    a.h[1] = *(const v8us*)(wrow + kb + 16 * half + 8);
    __builtin_prefetch(wrow + kb + 32, 0, 3);   // global_prefetch_b8

    const int kk = kb + 16 * half;              // B: 16 consecutive K per lane
    if constexpr (CIN % 16 == 0) {
      const int pos = kk / CIN;
      const int c0  = kk - pos * CIN;
      const int ky  = pos / 3;
      const int kx  = pos - ky * 3;
      const unsigned short* bp = smem + ((ky * 18 + m + kx) * CIN + c0);
      b0.h[0] = *(const v8us*)(bp);
      b0.h[1] = *(const v8us*)(bp + 8);
      b1.h[0] = *(const v8us*)(bp + 18 * CIN);       // row y0+1 = ky+1
      b1.h[1] = *(const v8us*)(bp + 18 * CIN + 8);
    } else {
      #pragma unroll
      for (int e = 0; e < 16; ++e) {
        int k    = kk + e;
        int pos  = k / CIN;
        int c    = k - pos * CIN;
        int ky   = pos / 3;
        int kx   = pos - ky * 3;
        int sidx = (k < K) ? ((ky * 18 + m + kx) * CIN + c) : 0;
        unsigned short v0 = smem[sidx];
        unsigned short v1 = smem[sidx + 18 * CIN];
        b0.u[e] = (k < K) ? v0 : (unsigned short)0;
        b1.u[e] = (k < K) ? v1 : (unsigned short)0;
      }
    }

    accA = __builtin_amdgcn_wmma_f32_16x16x32_bf16(
        false, a.v, false, b0.v, (short)0, accA, false, false);
    accB = __builtin_amdgcn_wmma_f32_16x16x32_bf16(
        false, a.v, false, b1.v, (short)0, accB, false, false);
  }

  // D layout: lane l, vgpr r -> (M = r + 8*(l>>4), N = l&15).
  // 2x2 pool: max over the two rows (accA/accB), then over x pairs via a
  // convergent lane^1 shuffle; bias+ReLU after (max commutes with both).
  // The 8 channels fb..fb+7 are contiguous in NHWC: pack to one b128 store.
  const int fb = f0 + 8 * half;
  const int xo = (xt + m) >> 1;
  const float* bp = bias + fb;         // 8 contiguous floats, vectorizable
  v8us outp;
  #pragma unroll
  for (int r = 0; r < 8; ++r) {
    float v = fmaxf(accA[r], accB[r]);
    v = fmaxf(v, __shfl_xor(v, 1, 32));
    float o = v + bp[r];
    o = o > 0.f ? o : 0.f;
    outp[r] = f2bf(o);
  }
  if ((m & 1) == 0) {
    *(v8us*)(P + (((size_t)n * HO + yo) * HO + xo) * F + fb) = outp;
  }
}

// FC: scores(64,10) = flat(64,32768) @ w_fc(32768,10) + b_fc (42 MFLOP).
// h3 is bf16 NHWC (64,16,16,128); reference flattening is NCHW, so
// d = c*256 + p with p = y*16+x. One block per image, one thread per pixel.
__global__ void fc_kernel(const unsigned short* __restrict__ h3,
                          const float* __restrict__ wfc,
                          const float* __restrict__ bfc,
                          float* __restrict__ out) {
  __shared__ float red[10][256];
  int n = blockIdx.x;
  int p = threadIdx.x;                 // pixel 0..255
  float acc[10];
  #pragma unroll
  for (int c = 0; c < 10; ++c) acc[c] = 0.f;
  const unsigned short* hp = h3 + ((size_t)n * 256 + p) * 128;
  for (int c = 0; c < 128; ++c) {
    float xv = bf2f(hp[c]);
    const float* wr = wfc + ((size_t)c * 256 + p) * 10;
    #pragma unroll
    for (int cls = 0; cls < 10; ++cls) acc[cls] += xv * wr[cls];
  }
  #pragma unroll
  for (int c = 0; c < 10; ++c) red[c][p] = acc[c];
  __syncthreads();
  for (int s = 128; s > 0; s >>= 1) {
    if (p < s) {
      #pragma unroll
      for (int c = 0; c < 10; ++c) red[c][p] += red[c][p + s];
    }
    __syncthreads();
  }
  if (p < 10) out[n * 10 + p] = red[p][0] + bfc[p];
}

extern "C" void kernel_launch(void* const* d_in, const int* in_sizes, int n_in,
                              void* d_out, int out_size, void* d_ws, size_t ws_size,
                              hipStream_t stream) {
  (void)in_sizes; (void)n_in; (void)out_size; (void)ws_size;

  const float* x   = (const float*)d_in[0];
  const float* w1  = (const float*)d_in[1];
  const float* b1  = (const float*)d_in[2];
  const float* w2  = (const float*)d_in[3];
  const float* b2  = (const float*)d_in[4];
  const float* w3  = (const float*)d_in[5];
  const float* b3  = (const float*)d_in[6];
  const float* wfc = (const float*)d_in[7];
  const float* bfc = (const float*)d_in[8];
  float* out = (float*)d_out;

  char* ws = (char*)d_ws;
  size_t off = 0;
  auto alloc = [&](size_t bytes) -> char* {
    char* p = ws + off;
    off = (off + bytes + 255) & ~(size_t)255;
    return p;
  };
  // Pooled activations, bf16 NHWC.
  unsigned short* h1 = (unsigned short*)alloc(64ull * 64 * 64 * 32 * 2);
  unsigned short* h2 = (unsigned short*)alloc(64ull * 32 * 32 * 64 * 2);
  unsigned short* h3 = (unsigned short*)alloc(64ull * 16 * 16 * 128 * 2);
  unsigned short* wp1 = (unsigned short*)alloc(32ull * 32 * 2);
  unsigned short* wp2 = (unsigned short*)alloc(64ull * 288 * 2);
  unsigned short* wp3 = (unsigned short*)alloc(128ull * 576 * 2);

  prep_weights_kernel<<<(32 * 32 + 255) / 256, 256, 0, stream>>>(w1, wp1, 32, 3, 32);
  prep_weights_kernel<<<(64 * 288 + 255) / 256, 256, 0, stream>>>(w2, wp2, 64, 32, 288);
  prep_weights_kernel<<<(128 * 576 + 255) / 256, 256, 0, stream>>>(w3, wp3, 128, 64, 576);

  // L1: 3->32, 128x128 -> pooled 64x64
  conv_pool_wmma<3, 32, 128, 2, false>
      <<<dim3(8, 64, 64), 64, 0, stream>>>(x, wp1, b1, h1);
  // L2: 32->64, 64x64 -> pooled 32x32
  conv_pool_wmma<32, 64, 64, 4, true>
      <<<dim3(4, 32, 64), 128, 0, stream>>>(h1, wp2, b2, h2);
  // L3: 64->128, 32x32 -> pooled 16x16
  conv_pool_wmma<64, 128, 32, 4, true>
      <<<dim3(2, 16, 128), 128, 0, stream>>>(h2, wp3, b3, h3);

  // FC
  fc_kernel<<<64, 256, 0, stream>>>(h3, wfc, bfc, out);
}
